// Block_72353019068975
// MI455X (gfx1250) — matrix operand order repaired and checked by reading, not compiled, not run on hardware
//
#include <hip/hip_runtime.h>

// ---------------------------------------------------------------------------
// ViT encoder block for MI455X (gfx1250, wave32, WMMA + Tensor Data Mover).
// GEMMs: bf16 WMMA (fp32 accumulate), tiles staged via TENSOR_LOAD_TO_LDS
// (double-buffered, overlapped with compute, tracked by TENSORcnt).
// ---------------------------------------------------------------------------

typedef __bf16 bf16;
typedef __attribute__((ext_vector_type(16))) bf16     v16bf;
typedef __attribute__((ext_vector_type(8)))  bf16     v8bf;
typedef __attribute__((ext_vector_type(8)))  float    v8f;
typedef __attribute__((ext_vector_type(4)))  unsigned u32x4;
typedef __attribute__((ext_vector_type(8)))  int      i32x8;
typedef __attribute__((ext_vector_type(4)))  int      i32x4;

#define DIMD   1024
#define HIDD   4096
#define TSEQ   577
#define NBATCH 32
#define BTROWS (NBATCH * TSEQ)   // 18464
#define NHEAD  16
#define HDIM   64
#define QKVLD  3072

// ---------------- small helpers ----------------

__device__ __forceinline__ v8f v8f_zero() {
  v8f z = {0.f, 0.f, 0.f, 0.f, 0.f, 0.f, 0.f, 0.f};
  return z;
}

__device__ __forceinline__ v8bf v8bf_zero() {
  v8bf z;
#pragma unroll
  for (int i = 0; i < 8; ++i) z[i] = (bf16)0.0f;
  return z;
}

__device__ __forceinline__ v16bf cat8(v8bf lo, v8bf hi) {
  v16bf r;
#pragma unroll
  for (int i = 0; i < 8; ++i) { r[i] = lo[i]; r[i + 8] = hi[i]; }
  return r;
}

__device__ __forceinline__ v8f wmma_bf16(v16bf a, v16bf b, v8f c) {
  return __builtin_amdgcn_wmma_f32_16x16x32_bf16(
      false, a, false, b, (short)0, c, false, false);
}

__device__ __forceinline__ float wave_red_sum(float v) {
#pragma unroll
  for (int off = 16; off > 0; off >>= 1) v += __shfl_xor(v, off, 32);
  return v;
}

// low 32 bits of a flat LDS pointer are the LDS byte offset (aperture layout)
__device__ __forceinline__ unsigned lds_byte_off(const void* p) {
  return (unsigned)(unsigned long long)p;
}

// ---------------- Tensor Data Mover: 2D tile load, bf16 elements ----------
// Loads a tile_dim0(=32 elems) x tile_dim1(=128 rows) tile from a row-major
// (rows x stride_elems) bf16 tensor into LDS. pad: +16B after every 64B row
// -> LDS row stride 80B (16B aligned, bank-conflict-free for fragment reads).
// OOB rows (y >= td1) are zero-filled by the TDM.

__device__ __forceinline__ void tdm_load_tile_2d(unsigned lds_off,
                                                 const bf16* gaddr,
                                                 unsigned td0, unsigned td1,
                                                 unsigned stride_elems) {
  unsigned long long ga = (unsigned long long)gaddr;
  u32x4 g0;
  g0[0] = 1u;                                   // count=1, user descriptor
  g0[1] = lds_off;                              // lds_addr (bytes)
  g0[2] = (unsigned)ga;                         // global_addr[31:0]
  g0[3] = (unsigned)(ga >> 32) | (2u << 30);    // global_addr[56:32] | type=2
  i32x8 g1;
  // data_size=1(2B) @16; pad_enable @20; pad_interval=3(64B) @22; pad_amount=3(16B) @25
  g1[0] = (int)((1u << 16) | (1u << 20) | (3u << 22) | (3u << 25));
  g1[1] = (int)((td0 & 0xFFFFu) << 16);                    // tensor_dim0[15:0]
  g1[2] = (int)((td0 >> 16) | ((td1 & 0xFFFFu) << 16));    // td0 hi | td1 lo
  g1[3] = (int)((td1 >> 16) | (32u << 16));                // td1 hi | tile_dim0=32
  g1[4] = (int)128;                                        // tile_dim1=128, tile_dim2=0
  g1[5] = (int)stride_elems;                               // tensor_dim0_stride lo
  g1[6] = 0;                                               // stride hi | dim1_stride lo
  g1[7] = 0;
  i32x4 z4 = {0, 0, 0, 0};
#if __clang_major__ >= 23
  i32x8 z8 = {0, 0, 0, 0, 0, 0, 0, 0};
  __builtin_amdgcn_tensor_load_to_lds(g0, g1, z4, z4, z8, 0);
#else
  __builtin_amdgcn_tensor_load_to_lds(g0, g1, z4, z4, 0);
#endif
}

// ---------------- weight transpose + fp32->bf16 -------------------------
// wt[n][k] = (bf16) w[k][n]; 32x32 LDS tiles, coalesced both sides.

__global__ __launch_bounds__(256) void transpose_cvt(const float* __restrict__ w,
                                                     bf16* __restrict__ wt,
                                                     int K, int N) {
  __shared__ float t[32][33];
  const int n0 = blockIdx.x * 32, k0 = blockIdx.y * 32;
  const int tx = threadIdx.x & 31, ty = threadIdx.x >> 5;
#pragma unroll
  for (int r = ty; r < 32; r += 8)
    t[r][tx] = w[(size_t)(k0 + r) * N + (n0 + tx)];
  __syncthreads();
#pragma unroll
  for (int r = ty; r < 32; r += 8)
    wt[(size_t)(n0 + r) * K + (k0 + tx)] = (bf16)t[tx][r];
}

// ---------------- fused LayerNorm (fp32 in, bf16 out) ----------------

__global__ __launch_bounds__(256) void ln_kernel(const float* __restrict__ x,
                                                 const float* __restrict__ g,
                                                 const float* __restrict__ b,
                                                 bf16* __restrict__ out) {
  const int row = blockIdx.x;
  const int tid = threadIdx.x;
  const float* xr = x + (size_t)row * DIMD;
  float4 v = ((const float4*)xr)[tid];
  float s  = v.x + v.y + v.z + v.w;
  float sq = v.x * v.x + v.y * v.y + v.z * v.z + v.w * v.w;
  s  = wave_red_sum(s);
  sq = wave_red_sum(sq);

  __shared__ float ls[8], lq[8];
  const int wid = tid >> 5, lane = tid & 31;
  if (lane == 0) { ls[wid] = s; lq[wid] = sq; }
  __syncthreads();
  if (tid == 0) {
    float S = 0.f, Q = 0.f;
#pragma unroll
    for (int i = 0; i < 8; ++i) { S += ls[i]; Q += lq[i]; }
    ls[0] = S; lq[0] = Q;
  }
  __syncthreads();
  const float mu  = ls[0] * (1.f / (float)DIMD);
  const float var = lq[0] * (1.f / (float)DIMD) - mu * mu;
  const float rs  = rsqrtf(var + 1e-6f);

  float4 gg = ((const float4*)g)[tid];
  float4 bb = ((const float4*)b)[tid];
  bf16* orow = out + (size_t)row * DIMD + tid * 4;
  orow[0] = (bf16)((v.x - mu) * rs * gg.x + bb.x);
  orow[1] = (bf16)((v.y - mu) * rs * gg.y + bb.y);
  orow[2] = (bf16)((v.z - mu) * rs * gg.z + bb.z);
  orow[3] = (bf16)((v.w - mu) * rs * gg.w + bb.w);
}

// ---------------- bf16 WMMA GEMM, TDM-staged, double-buffered ------------
// C(M,N) = A(M,K) * Bt(N,K)^T  [+bias][gelu][+res]
// tile 128x128x32; 8 waves in 2(M)x4(N); wave = 64x32 via 4x2 accumulators.
// Wave 0 drives the TDM; stage i+1 overlaps WMMA on stage i.

template <bool DO_GELU, bool DO_RES, bool OUT_BF>
__global__ __launch_bounds__(256)
void gemm_kernel(const bf16* __restrict__ A, int lda,
                 const bf16* __restrict__ Bt, int ldb,   // (N,K) row-major, ldb==K
                 const float* __restrict__ bias,
                 const float* __restrict__ res,
                 float* __restrict__ outF, bf16* __restrict__ outB,
                 int M, int N, int K) {
  __shared__ bf16 As[2][128][40];  // TDM pads 64B rows to 80B stride
  __shared__ bf16 Bs[2][128][40];

  const int tid = threadIdx.x;
  const int n0 = blockIdx.x * 128;
  const int m0 = blockIdx.y * 128;
  const int wid = tid >> 5, lane = tid & 31;
  const int wm = wid >> 2, wn = wid & 3;
  const int l16 = lane & 15, kh = lane >> 4;

  const unsigned offA0 = lds_byte_off(&As[0][0][0]);
  const unsigned offA1 = lds_byte_off(&As[1][0][0]);
  const unsigned offB0 = lds_byte_off(&Bs[0][0][0]);
  const unsigned offB1 = lds_byte_off(&Bs[1][0][0]);

  const bf16* Ab = A  + (size_t)m0 * lda;
  const bf16* Bb = Bt + (size_t)n0 * ldb;
  const unsigned mrem = (unsigned)(M - m0);
  const unsigned nrem = (unsigned)(N - n0);

  v8f acc[4][2];
#pragma unroll
  for (int mt = 0; mt < 4; ++mt)
#pragma unroll
    for (int nt = 0; nt < 2; ++nt) acc[mt][nt] = v8f_zero();

  // prologue: stage k0 = 0 into buffer 0
  if (wid == 0) {
    tdm_load_tile_2d(offA0, Ab, (unsigned)K, mrem, (unsigned)lda);
    tdm_load_tile_2d(offB0, Bb, (unsigned)K, nrem, (unsigned)ldb);
  }

  int buf = 0;
  for (int k0 = 0; k0 < K; k0 += 32) {
    if (wid == 0) __builtin_amdgcn_s_wait_tensorcnt(0);
    __syncthreads();  // tile[buf] visible to all; tile[buf^1] fully consumed
    if (wid == 0 && (k0 + 32) < K) {
      const unsigned a1 = buf ? offA0 : offA1;
      const unsigned b1 = buf ? offB0 : offB1;
      tdm_load_tile_2d(a1, Ab + (k0 + 32), (unsigned)(K - k0 - 32), mrem, (unsigned)lda);
      tdm_load_tile_2d(b1, Bb + (k0 + 32), (unsigned)(K - k0 - 32), nrem, (unsigned)ldb);
    }

    v16bf af[4];
#pragma unroll
    for (int mt = 0; mt < 4; ++mt) {
      const int r = wm * 64 + mt * 16 + l16;
      af[mt] = cat8(*(const v8bf*)&As[buf][r][kh * 8],
                    *(const v8bf*)&As[buf][r][16 + kh * 8]);
    }
    v16bf bfv[2];
#pragma unroll
    for (int nt = 0; nt < 2; ++nt) {
      const int cn = wn * 32 + nt * 16 + l16;
      bfv[nt] = cat8(*(const v8bf*)&Bs[buf][cn][kh * 8],
                     *(const v8bf*)&Bs[buf][cn][16 + kh * 8]);
    }
#pragma unroll
    for (int mt = 0; mt < 4; ++mt)
#pragma unroll
      for (int nt = 0; nt < 2; ++nt)
        acc[mt][nt] = wmma_bf16(af[mt], bfv[nt], acc[mt][nt]);

    buf ^= 1;
  }

  // epilogue
#pragma unroll
  for (int mt = 0; mt < 4; ++mt) {
#pragma unroll
    for (int nt = 0; nt < 2; ++nt) {
#pragma unroll
      for (int i = 0; i < 8; ++i) {
        const int gm = m0 + wm * 64 + mt * 16 + kh * 8 + i;
        const int gn = n0 + wn * 32 + nt * 16 + l16;
        if (gm < M) {
          float v = acc[mt][nt][i] + bias[gn];
          if (DO_GELU) v = 0.5f * v * (1.f + erff(v * 0.70710678118654752f));
          const size_t idx = (size_t)gm * N + gn;
          if (DO_RES) v += res[idx];
          if (OUT_BF) outB[idx] = (bf16)v;
          else        outF[idx] = v;
        }
      }
    }
  }
}

// ---------------- fused attention ----------------
// grid: (ceil(T/64), B*H). One WG = 64 queries of one (b,h).
// S(64x608) fp32 in LDS; K fragments straight from global (d-contiguous
// rows match the WMMA B layout); V staged transposed; softmax in place.

#define TPADK 608                 // 19 * 32
#define SSTR  609                 // odd stride -> conflict-free column reads
#define QSTR  72
#define VSTR  616
#define S_BYTES  (64 * SSTR * 4)  // 155904
#define Q_BYTES  (64 * QSTR * 2)  // 9216
#define V_BYTES  (64 * VSTR * 2)  // 78848
#define ATTN_SMEM (S_BYTES + Q_BYTES + V_BYTES + 2304)

__global__ __launch_bounds__(256)
void attn_kernel(const bf16* __restrict__ qkv, bf16* __restrict__ out) {
  extern __shared__ char smem[];
  float* S    = (float*)smem;
  bf16*  Qs   = (bf16*)(smem + S_BYTES);
  bf16*  Vt   = (bf16*)(smem + S_BYTES + Q_BYTES);
  float* red  = (float*)(smem + S_BYTES + Q_BYTES + V_BYTES);
  float* red2 = red + 256;
  float* rsc  = red + 512;

  const int tid = threadIdx.x;
  const int q0  = blockIdx.x * 64;
  const int bh  = blockIdx.y;
  const int bb  = bh >> 4, hh = bh & 15;
  const size_t base = (size_t)(bb * TSEQ) * QKVLD;
  const bf16* Qg = qkv + base + hh * HDIM;
  const bf16* Kg = qkv + base + DIMD + hh * HDIM;
  const bf16* Vg = qkv + base + 2 * DIMD + hh * HDIM;

  for (int c = tid; c < 512; c += 256) {
    const int r = c >> 3, d0 = (c & 7) * 8;
    const int q = q0 + r;
    v8bf v = (q < TSEQ) ? *(const v8bf*)(Qg + (size_t)q * QKVLD + d0) : v8bf_zero();
    *(v8bf*)&Qs[r * QSTR + d0] = v;
  }
  for (int c = tid; c < TPADK * 8; c += 256) {
    const int t = c >> 3, d0 = (c & 7) * 8;
    v8bf v = (t < TSEQ) ? *(const v8bf*)(Vg + (size_t)t * QKVLD + d0) : v8bf_zero();
#pragma unroll
    for (int j = 0; j < 8; ++j) Vt[(d0 + j) * VSTR + t] = v[j];
  }
  __syncthreads();

  const int wid = tid >> 5, lane = tid & 31;
  const int l16 = lane & 15, kh = lane >> 4;

  v16bf qf[4][2];
#pragma unroll
  for (int mt = 0; mt < 4; ++mt)
#pragma unroll
    for (int kk = 0; kk < 2; ++kk) {
      const int r = mt * 16 + l16;
      const int d = kk * 32 + kh * 8;
      qf[mt][kk] = cat8(*(const v8bf*)&Qs[r * QSTR + d],
                        *(const v8bf*)&Qs[r * QSTR + d + 16]);
    }

  for (int ntile = wid; ntile < TPADK / 16; ntile += 8) {
    v8f acc[4];
#pragma unroll
    for (int mt = 0; mt < 4; ++mt) acc[mt] = v8f_zero();
    const int t = ntile * 16 + l16;
    const bool tin = (t < TSEQ);
    const bf16* kprow = Kg + (size_t)t * QKVLD;
#pragma unroll
    for (int kk = 0; kk < 2; ++kk) {
      const int d = kk * 32 + kh * 8;
      v8bf lo, hi;
      if (tin) { lo = *(const v8bf*)(kprow + d); hi = *(const v8bf*)(kprow + d + 16); }
      else     { lo = v8bf_zero();               hi = v8bf_zero(); }
      const v16bf bfr = cat8(lo, hi);
#pragma unroll
      for (int mt = 0; mt < 4; ++mt) acc[mt] = wmma_bf16(qf[mt][kk], bfr, acc[mt]);
    }
#pragma unroll
    for (int mt = 0; mt < 4; ++mt)
#pragma unroll
      for (int i = 0; i < 8; ++i)
        S[(mt * 16 + kh * 8 + i) * SSTR + ntile * 16 + l16] = acc[mt][i] * 0.125f;
  }
  __syncthreads();

  {
    const int row = tid >> 2, part = tid & 3;
    const int c0 = part * 152, c1 = c0 + 152;
    float m = -1e30f;
    for (int c = c0; c < c1 && c < TSEQ; ++c) m = fmaxf(m, S[row * SSTR + c]);
    red[row * 4 + part] = m;
    __syncthreads();
    const float rm = fmaxf(fmaxf(red[row * 4 + 0], red[row * 4 + 1]),
                           fmaxf(red[row * 4 + 2], red[row * 4 + 3]));
    float sum = 0.f;
    for (int c = c0; c < c1; ++c) {
      if (c < TSEQ) {
        const float e = __expf(S[row * SSTR + c] - rm);
        S[row * SSTR + c] = e;
        sum += e;
      } else {
        S[row * SSTR + c] = 0.f;
      }
    }
    red2[row * 4 + part] = sum;
    __syncthreads();
    if (part == 0) {
      const float l = red2[row * 4 + 0] + red2[row * 4 + 1] +
                      red2[row * 4 + 2] + red2[row * 4 + 3];
      rsc[row] = 1.f / l;
    }
  }
  __syncthreads();

  const int mt = wid >> 1, nh = wid & 1;
  v8f o[2];
  o[0] = v8f_zero(); o[1] = v8f_zero();
  const int mrow = mt * 16 + l16;
  for (int t0 = 0; t0 < TPADK; t0 += 32) {
    v16bf af;
#pragma unroll
    for (int v = 0; v < 8; ++v) {
      const int k = t0 + ((v >> 2) << 4) + (kh << 3) + ((v & 3) << 1);
      af[2 * v]     = (bf16)S[mrow * SSTR + k];
      af[2 * v + 1] = (bf16)S[mrow * SSTR + k + 1];
    }
#pragma unroll
    for (int nt = 0; nt < 2; ++nt) {
      const int d = nh * 32 + nt * 16 + l16;
      const v16bf bfr = cat8(*(const v8bf*)&Vt[d * VSTR + t0 + kh * 8],
                             *(const v8bf*)&Vt[d * VSTR + t0 + 16 + kh * 8]);
      o[nt] = wmma_bf16(af, bfr, o[nt]);
    }
  }
#pragma unroll
  for (int nt = 0; nt < 2; ++nt)
#pragma unroll
    for (int i = 0; i < 8; ++i) {
      const int r = mt * 16 + kh * 8 + i;
      const int q = q0 + r;
      if (q < TSEQ) {
        const int d = nh * 32 + nt * 16 + l16;
        out[(size_t)(bb * TSEQ + q) * DIMD + hh * HDIM + d] = (bf16)(o[nt][i] * rsc[r]);
      }
    }
}

// ---------------- host launcher ----------------

extern "C" void kernel_launch(void* const* d_in, const int* in_sizes, int n_in,
                              void* d_out, int out_size, void* d_ws, size_t ws_size,
                              hipStream_t stream) {
  (void)in_sizes; (void)n_in; (void)out_size; (void)ws_size;
  const float* x      = (const float*)d_in[0];
  const float* ln1_g  = (const float*)d_in[1];
  const float* ln1_b  = (const float*)d_in[2];
  const float* qkv_w  = (const float*)d_in[3];
  const float* qkv_b  = (const float*)d_in[4];
  const float* proj_w = (const float*)d_in[5];
  const float* proj_b = (const float*)d_in[6];
  const float* ln2_g  = (const float*)d_in[7];
  const float* ln2_b  = (const float*)d_in[8];
  const float* fc1_w  = (const float*)d_in[9];
  const float* fc1_b  = (const float*)d_in[10];
  const float* fc2_w  = (const float*)d_in[11];
  const float* fc2_b  = (const float*)d_in[12];
  float* outF = (float*)d_out;  // also holds x1 (post-attention residual)

  char* ws = (char*)d_ws;
  const size_t o_wqkv = 0;
  const size_t o_wprj = o_wqkv + (size_t)1024 * 3072 * 2;
  const size_t o_wfc1 = o_wprj + (size_t)1024 * 1024 * 2;
  const size_t o_wfc2 = o_wfc1 + (size_t)1024 * 4096 * 2;
  const size_t o_xn   = o_wfc2 + (size_t)4096 * 1024 * 2;
  const size_t o_qkv  = o_xn   + (size_t)BTROWS * DIMD * 2;
  const size_t o_attn = o_qkv  + (size_t)BTROWS * QKVLD * 2;
  const size_t o_h    = o_qkv;  // h (BTROWS x 4096 bf16) aliases dead qkv+attn

  bf16* w_qkvT = (bf16*)(ws + o_wqkv);   // (3072, 1024)
  bf16* w_prjT = (bf16*)(ws + o_wprj);   // (1024, 1024)
  bf16* w_fc1T = (bf16*)(ws + o_wfc1);   // (4096, 1024)
  bf16* w_fc2T = (bf16*)(ws + o_wfc2);   // (1024, 4096)
  bf16* xn    = (bf16*)(ws + o_xn);
  bf16* qkvb  = (bf16*)(ws + o_qkv);
  bf16* attnb = (bf16*)(ws + o_attn);
  bf16* hb    = (bf16*)(ws + o_h);

  // weight transpose + fp32->bf16 (once per call; reused across 145 M-tiles)
  transpose_cvt<<<dim3(3072 / 32, 1024 / 32), 256, 0, stream>>>(qkv_w,  w_qkvT, 1024, 3072);
  transpose_cvt<<<dim3(1024 / 32, 1024 / 32), 256, 0, stream>>>(proj_w, w_prjT, 1024, 1024);
  transpose_cvt<<<dim3(4096 / 32, 1024 / 32), 256, 0, stream>>>(fc1_w,  w_fc1T, 1024, 4096);
  transpose_cvt<<<dim3(1024 / 32, 4096 / 32), 256, 0, stream>>>(fc2_w,  w_fc2T, 4096, 1024);

  // LN1
  ln_kernel<<<BTROWS, 256, 0, stream>>>(x, ln1_g, ln1_b, xn);

  // QKV GEMM -> bf16
  gemm_kernel<false, false, true><<<dim3(3072 / 128, (BTROWS + 127) / 128), 256, 0, stream>>>(
      xn, DIMD, w_qkvT, DIMD, qkv_b, nullptr, nullptr, qkvb, BTROWS, 3072, DIMD);

  // fused attention
  (void)hipFuncSetAttribute(reinterpret_cast<const void*>(attn_kernel),
                            hipFuncAttributeMaxDynamicSharedMemorySize,
                            (int)ATTN_SMEM);
  attn_kernel<<<dim3((TSEQ + 63) / 64, NBATCH * NHEAD), 256, ATTN_SMEM, stream>>>(qkvb, attnb);

  // proj GEMM + residual(x) -> x1 in d_out (fp32)
  gemm_kernel<false, true, false><<<dim3(DIMD / 128, (BTROWS + 127) / 128), 256, 0, stream>>>(
      attnb, DIMD, w_prjT, DIMD, proj_b, x, outF, nullptr, BTROWS, DIMD, DIMD);

  // LN2
  ln_kernel<<<BTROWS, 256, 0, stream>>>(outF, ln2_g, ln2_b, xn);

  // FC1 GEMM + bias + exact GELU -> bf16 h
  gemm_kernel<true, false, true><<<dim3(HIDD / 128, (BTROWS + 127) / 128), 256, 0, stream>>>(
      xn, DIMD, w_fc1T, DIMD, fc1_b, nullptr, nullptr, hb, BTROWS, HIDD, DIMD);

  // FC2 GEMM + bias + residual(d_out) -> d_out (same-index RMW per thread)
  gemm_kernel<false, true, false><<<dim3(DIMD / 128, (BTROWS + 127) / 128), 256, 0, stream>>>(
      hb, HIDD, w_fc2T, HIDD, fc2_b, outF, outF, nullptr, BTROWS, DIMD, HIDD);
}
